// Net_22634477650649
// MI455X (gfx1250) — matrix operand order repaired and checked by reading, not compiled, not run on hardware
//
#include <hip/hip_runtime.h>
#include <hip/hip_bf16.h>

typedef __attribute__((ext_vector_type(16))) _Float16 v16h;
typedef __attribute__((ext_vector_type(8)))  float    v8f;
typedef __attribute__((ext_vector_type(2)))  __fp16   fp16x2;
typedef int v4i_b128 __attribute__((vector_size(4 * sizeof(int))));

#define NODES 128
#define FIN   768
#define FMID  16
#define AS    136            // sA row stride (halves), padded vs bank conflicts
#define XS    68             // f32 x-staging row stride (floats), 272B = 16B-aligned rows

#if __has_builtin(__builtin_amdgcn_global_load_async_to_lds_b128) && \
    __has_builtin(__builtin_amdgcn_s_wait_asynccnt)
#define HAVE_ASYNC_LDS 1
#else
#define HAVE_ASYNC_LDS 0
#endif

// ---- LDS layout (byte offsets into dynamic shared) ------------------------
#define OFF_A    0                     // f16 [128][136]  A matrix      34816 B (persistent)
#define OFF_H1   34816                 // f16 [128][32]   y1, K-padded   8192 B (persistent)
#define OFF_DINV 43008                 // f32 [128]                       512 B
#define OFF_DEG  43520                 // i32 [128]                       512 B
#define OFF_RED  44032                 // f32 [256]  softmax partials    1024 B
#define OFF_U    45056                 // union region
//   layer-1 view:
#define OFF_W1   (OFF_U)               // f16 [16][768]   W1^T          24576 B
#define OFF_X0   (OFF_U + 24576)       // f32 [128][68]   x buf 0       34816 B
#define OFF_X1   (OFF_U + 59392)       // f32 [128][68]   x buf 1       34816 B
#define OFF_HX   (OFF_U + 94208)       // f16 [16][132]   Hx^T           4224 B
//   layer-2 view:
#define OFF_T    (OFF_U)               // f16 [128][136]  T^T           34816 B
#define OFF_W2   (OFF_U + 34816)       // f16 [128][34]   W2c^T          8704 B
#define OFF_OUT  (OFF_U + 43520)       // f32 [128][132]  out chunk     67584 B
#define SMEM_BYTES (OFF_U + 111104)    // = 156160 B (< 160KB -> 2 WGs per WGP)

__device__ __forceinline__ unsigned pkrtz(float a, float b) {
  union { fp16x2 h; unsigned u; } c;
  c.h = __builtin_amdgcn_cvt_pkrtz(a, b);   // one v_cvt_pk_rtz_f16_f32
  return c.u;
}

// A-operand gather (16x32 f16, ISA 7.12.2) from f16 storage:
// lanes 0-15: M=0..15, K={0..7,16..23}; lanes 16-31: same M, K={8..15,24..31}.
__device__ __forceinline__ v16h loadA16(const _Float16* base, int strideH,
                                        int row, int k0, int hi) {
  const _Float16* p = base + row * strideH + k0 + (hi ? 8 : 0);
  union { v16h v; unsigned u[8]; } r;
#pragma unroll
  for (int q = 0; q < 4; ++q) r.u[q] = *(const unsigned*)(p + 2 * q);
#pragma unroll
  for (int q = 4; q < 8; ++q) r.u[q] = *(const unsigned*)(p + 8 + 2 * q);
  return r.v;
}

// A-operand gather from raw f32 storage with on-the-fly f32->f16 packing.
__device__ __forceinline__ v16h loadA16cvt(const float* base, int strideF,
                                           int row, int k0, int hi) {
  const float* p = base + row * strideF + k0 + (hi ? 8 : 0);
  union { v16h v; unsigned u[8]; } r;
#pragma unroll
  for (int q = 0; q < 4; ++q) {
    float2 f = *(const float2*)(p + 2 * q);
    r.u[q] = pkrtz(f.x, f.y);
  }
#pragma unroll
  for (int q = 4; q < 8; ++q) {
    float2 f = *(const float2*)(p + 8 + 2 * q);
    r.u[q] = pkrtz(f.x, f.y);
  }
  return r.v;
}

// B-operand gather (32x16 f16) from transposed storage baseT[n][k]:
// lane n = nBase+(lane&15); lanes 0-15 K={0..7,16..23}, lanes 16-31 K={8..15,24..31}.
__device__ __forceinline__ v16h loadB16(const _Float16* baseT, int strideH,
                                        int n, int k0, int hi) {
  const _Float16* p = baseT + n * strideH + k0 + (hi ? 8 : 0);
  union { v16h v; unsigned u[8]; } r;
#pragma unroll
  for (int q = 0; q < 8; ++q) r.u[q] = *(const unsigned*)(p + 2 * q + ((q & 4) ? 8 : 0));
  return r.v;
}

#if HAVE_ASYNC_LDS
// Issue one 128x64-f32 x chunk as 8 async b128 transfers per thread (ASYNCcnt).
__device__ __forceinline__ void issue_chunk(const float* xg, float* buf,
                                            int kc, int tid) {
#pragma unroll
  for (int j = 0; j < 8; ++j) {
    int e = (tid + j * 256) << 2;           // f32 index within chunk
    int node = e >> 6, k = e & 63;
    __builtin_amdgcn_global_load_async_to_lds_b128(
        (__attribute__((address_space(1))) v4i_b128*)(xg + (size_t)node * FIN + kc * 64 + k),
        (__attribute__((address_space(3))) v4i_b128*)(buf + node * XS + k),
        0, 0);
  }
}
#endif

// One layer-1 K step: 2x WMMA over a staged 64-wide f32 chunk.
__device__ __forceinline__ v8f l1_step(v8f acc, const float* buf,
                                       const _Float16* sW1t, int kc,
                                       int row, int hi, int ln) {
  v16h a0 = loadA16cvt(buf, XS, row, 0, hi);
  v16h b0 = loadB16(sW1t, FIN, ln, kc * 64, hi);
  acc = __builtin_amdgcn_wmma_f32_16x16x32_f16(false, a0, false, b0,
                                               (short)0, acc, false, false);
  v16h a1 = loadA16cvt(buf, XS, row, 32, hi);
  v16h b1 = loadB16(sW1t, FIN, ln, kc * 64 + 32, hi);
  acc = __builtin_amdgcn_wmma_f32_16x16x32_f16(false, a1, false, b1,
                                               (short)0, acc, false, false);
  return acc;
}

__global__ __launch_bounds__(256, 1) void gcn2_wmma_kernel(
    const int* __restrict__ head, const float* __restrict__ x,
    const float* __restrict__ W1, const float* __restrict__ b1,
    const float* __restrict__ W2, const float* __restrict__ b2,
    float* __restrict__ out)
{
  extern __shared__ char smem[];
  _Float16* sA    = (_Float16*)(smem + OFF_A);
  _Float16* sH1   = (_Float16*)(smem + OFF_H1);
  float*    sDinv = (float*)(smem + OFF_DINV);
  int*      sDeg  = (int*)(smem + OFF_DEG);
  float*    sRed  = (float*)(smem + OFF_RED);
  _Float16* sW1t  = (_Float16*)(smem + OFF_W1);
  _Float16* sHxt  = (_Float16*)(smem + OFF_HX);
  _Float16* sTt   = (_Float16*)(smem + OFF_T);
  _Float16* sW2t  = (_Float16*)(smem + OFF_W2);
  float*    sOut  = (float*)(smem + OFF_OUT);
  float*    xbuf0 = (float*)(smem + OFF_X0);
  float*    xbuf1 = (float*)(smem + OFF_X1);

  const int g    = blockIdx.x;
  const int tid  = threadIdx.x;
  const int lane = tid & 31;
  const int wv   = tid >> 5;            // wave id 0..7 -> M-tile
  const int hi   = (lane >> 4) & 1;
  const int ln   = lane & 15;

  // ---- Stage W1 transposed: sW1t[n*768 + k] = W1[k][n]
#pragma unroll
  for (int j = 0; j < (FIN * FMID) / 256; ++j) {
    int i = tid + j * 256;
    sW1t[(i & 15) * FIN + (i >> 4)] = (_Float16)W1[i];
  }

  // ---- degrees: deg[j] = 1 (self loop) + #{i: head[i]==j}; dinv = deg^-1/2
  if (tid < NODES) sDeg[tid] = 1;
  __syncthreads();
  if (tid < NODES) atomicAdd(&sDeg[head[g * NODES + tid]], 1);
  __syncthreads();
  if (tid < NODES) sDinv[tid] = rsqrtf((float)sDeg[tid]);
  __syncthreads();

  // ---- Dense normalized adjacency A (f16); column i owned by thread i.
#pragma unroll
  for (int j = 0; j < (NODES * AS / 2) / 256; ++j)
    ((unsigned*)sA)[tid + j * 256] = 0u;
  __syncthreads();
  if (tid < NODES) {
    float di = sDinv[tid];
    sA[tid * AS + tid] = (_Float16)(di * di);                   // self loop
    int h = head[g * NODES + tid];
    float v = di * sDinv[h];
    sA[h * AS + tid] = (_Float16)((float)sA[h * AS + tid] + v); // edge i->head[i]
  }
  __syncthreads();

  // ============================ LAYER 1 =====================================
  // Hx = x @ W1 (128x768x16). x staged raw f32 (b128 copies), converted to
  // f16 inside the operand gather (each element is consumed exactly once).
  v8f accH = {};
  const float* xg = x + (size_t)g * NODES * FIN;
#if HAVE_ASYNC_LDS
  // Double-buffered async global->LDS (ASYNCcnt), overlapped with WMMA.
  issue_chunk(xg, xbuf0, 0, tid);
  issue_chunk(xg, xbuf1, 1, tid);
  __builtin_amdgcn_s_wait_asynccnt(8);      // chunk 0 landed (in-order)
  __syncthreads();
  for (int kc = 0; kc < FIN / 64; ++kc) {
    accH = l1_step(accH, (kc & 1) ? xbuf1 : xbuf0, sW1t, kc, wv * 16 + ln, hi, ln);
    __syncthreads();                        // all waves done reading buf
    if (kc + 2 < FIN / 64) {
      issue_chunk(xg, (kc & 1) ? xbuf1 : xbuf0, kc + 2, tid);
      __builtin_amdgcn_s_wait_asynccnt(8);  // chunk kc+1 landed
    } else {
      __builtin_amdgcn_s_wait_asynccnt(0);
    }
    __syncthreads();                        // landed data visible to all waves
  }
#else
  for (int kc = 0; kc < FIN / 64; ++kc) {
#pragma unroll
    for (int j = 0; j < 8; ++j) {           // raw b128 copy, no convert VALU
      int e = (tid + j * 256) << 2;
      int node = e >> 6, k = e & 63;
      float4 xv = *(const float4*)(xg + (size_t)node * FIN + kc * 64 + k);
      *(float4*)(xbuf0 + node * XS + k) = xv;
      if (kc + 1 < FIN / 64)                // global_prefetch_b8 next chunk
        __builtin_prefetch(xg + (size_t)node * FIN + (kc + 1) * 64 + k, 0, 1);
    }
    __syncthreads();
    accH = l1_step(accH, xbuf0, sW1t, kc, wv * 16 + ln, hi, ln);
    __syncthreads();
  }
#endif
  // store Hx transposed (B-role): sHxt[n*132 + m]
#pragma unroll
  for (int p = 0; p < 4; ++p)
    *(unsigned*)(sHxt + ln * 132 + wv * 16 + hi * 8 + 2 * p) =
        pkrtz(accH[2 * p], accH[2 * p + 1]);
  __syncthreads();

  // y1 = relu(A @ Hx + b1) -> sH1 (A-role [128][32], K-cols 16..31 zero)
  {
    v8f acc = {};
#pragma unroll
    for (int kc = 0; kc < 4; ++kc) {
      v16h a = loadA16(sA, AS, wv * 16 + ln, kc * 32, hi);
      v16h b = loadB16(sHxt, 132, ln, kc * 32, hi);
      acc = __builtin_amdgcn_wmma_f32_16x16x32_f16(false, a, false, b,
                                                   (short)0, acc, false, false);
    }
    float b1v = b1[ln];
#pragma unroll
    for (int v = 0; v < 8; ++v) {
      int m = wv * 16 + hi * 8 + v;
      float r = fmaxf(acc[v] + b1v, 0.0f);
      sH1[m * 32 + ln]      = (_Float16)r;
      sH1[m * 32 + 16 + ln] = (_Float16)0.0f;   // K pad
    }
  }
  __syncthreads();

  // ============================ LAYER 2 =====================================
  // out = log_softmax(A @ (y1 @ W2) + b2), streamed 128 features per chunk.
  for (int c = 0; c < FIN / 128; ++c) {
    // stage W2[:, c*128:+128] transposed + K-padded: sW2t[n*34 + k]
#pragma unroll
    for (int j = 0; j < 5; ++j) {               // zero pad k = 16..33
      int i = tid + j * 256;
      if (i < 128 * 9) {
        int n = i / 9, kk = (i % 9) * 2 + 16;
        *(unsigned*)(sW2t + n * 34 + kk) = 0u;
      }
    }
#pragma unroll
    for (int j = 0; j < (FMID * 128) / 256; ++j) {
      int i = tid + j * 256;
      int k = i >> 7, n = i & 127;
      sW2t[n * 34 + k] = (_Float16)W2[k * FIN + c * 128 + n];
    }
    __syncthreads();

    // T = y1 @ W2c (128x32x128) -> f16 transposed in sTt[nfeat*136 + mnode]
    {
      v16h ay = loadA16(sH1, 32, wv * 16 + ln, 0, hi);
#pragma unroll
      for (int nt = 0; nt < 8; ++nt) {
        v16h b = loadB16(sW2t, 34, nt * 16 + ln, 0, hi);
        v8f t = {};
        t = __builtin_amdgcn_wmma_f32_16x16x32_f16(false, ay, false, b,
                                                   (short)0, t, false, false);
        int n = nt * 16 + ln;
#pragma unroll
        for (int p = 0; p < 4; ++p)
          *(unsigned*)(sTt + n * 136 + wv * 16 + hi * 8 + 2 * p) =
              pkrtz(t[2 * p], t[2 * p + 1]);
      }
    }
    __syncthreads();

    // Out_c = A @ T + b2 (128x128x128); 8 N-tile accumulators in VGPRs so
    // each A operand load is reused 8x.
    {
      v8f o[8];
      v8f z = {};
#pragma unroll
      for (int nt = 0; nt < 8; ++nt) o[nt] = z;
#pragma unroll
      for (int kc = 0; kc < 4; ++kc) {
        v16h a = loadA16(sA, AS, wv * 16 + ln, kc * 32, hi);
#pragma unroll
        for (int nt = 0; nt < 8; ++nt) {
          v16h b = loadB16(sTt, 136, nt * 16 + ln, kc * 32, hi);
          o[nt] = __builtin_amdgcn_wmma_f32_16x16x32_f16(false, a, false, b,
                                                         (short)0, o[nt],
                                                         false, false);
        }
      }
#pragma unroll
      for (int nt = 0; nt < 8; ++nt) {
        float b2v = b2[c * 128 + nt * 16 + ln];
#pragma unroll
        for (int v = 0; v < 8; ++v)
          sOut[(wv * 16 + hi * 8 + v) * 132 + nt * 16 + ln] = o[nt][v] + b2v;
      }
    }
    __syncthreads();

    // log_softmax over nodes (rows) per feature column, all 256 threads:
    // each (half, col) pair reduces 64 rows, halves combined via sRed.
    {
      int col = tid & 127;
      int hf  = tid >> 7;
      int r0  = hf * 64;
      float mx = -3.402823466e38f;
      for (int r = r0; r < r0 + 64; ++r) mx = fmaxf(mx, sOut[r * 132 + col]);
      sRed[hf * 128 + col] = mx;
      __syncthreads();
      float gmx = fmaxf(sRed[col], sRed[128 + col]);
      float s = 0.0f;
      for (int r = r0; r < r0 + 64; ++r) s += __expf(sOut[r * 132 + col] - gmx);
      __syncthreads();
      sRed[hf * 128 + col] = s;
      __syncthreads();
      float lse = gmx + __logf(sRed[col] + sRed[128 + col]);
      float* og = out + (size_t)g * NODES * FIN + c * 128 + col;
      for (int r = r0; r < r0 + 64; ++r)
        og[(size_t)r * FIN] = sOut[r * 132 + col] - lse;
    }
    __syncthreads();
  }
}

extern "C" void kernel_launch(void* const* d_in, const int* in_sizes, int n_in,
                              void* d_out, int out_size, void* d_ws, size_t ws_size,
                              hipStream_t stream) {
  const int*   head = (const int*)d_in[0];
  const float* x    = (const float*)d_in[1];
  const float* W1   = (const float*)d_in[2];
  const float* b1   = (const float*)d_in[3];
  const float* W2   = (const float*)d_in[4];
  const float* b2   = (const float*)d_in[5];
  float* out = (float*)d_out;
  (void)d_ws; (void)ws_size; (void)n_in; (void)out_size;

  const int B = in_sizes[0] / NODES;   // 256 graphs, one workgroup each
  hipLaunchKernelGGL(gcn2_wmma_kernel, dim3(B), dim3(256), SMEM_BYTES, stream,
                     head, x, W1, b1, W2, b2, out);
}
